// GroupedGRULayerTF_18631568130455
// MI455X (gfx1250) — compile-verified, matrix operand order checked by reading, not run on hardware
//
#include <hip/hip_runtime.h>
#include <math.h>

// Grouped GRU, TF gate order [z,r,n], reset_after=True.
// G=8 groups, B=32, T=1000, Ig=Hg=64, 3Hg=192, I_TOT=H_TOT=512.
// One persistent workgroup (256 threads = 8 wave32) per group.
// W/U kept in registers as bf16 WMMA B-fragments for all 1000 steps.

#define G_   8
#define B_   32
#define T_   1000
#define IG_  64
#define HG_  64
#define N3_  192
#define ITOT_ 512
#define GXS_ 196   // padded row stride for gx/gh in LDS (bank-conflict avoidance)

typedef __attribute__((ext_vector_type(16))) __bf16 v16bf;
typedef __attribute__((ext_vector_type(8)))  float  v8f;

__global__ __launch_bounds__(256, 1)
void grouped_gru_persistent(const float* __restrict__ x,     // [B,T,512]
                            const float* __restrict__ h0,    // [G,B,64]
                            const float* __restrict__ Wm,    // [G,64,192]
                            const float* __restrict__ Um,    // [G,64,192]
                            const float* __restrict__ bi,    // [G,192]
                            const float* __restrict__ bh,    // [G,192]
                            float* __restrict__ y,           // [B,T,512]
                            float* __restrict__ hout)        // [G,B,64]
{
    __shared__ float s_gx[B_][GXS_];                 // x_t · W   (f32 accum)
    __shared__ float s_gh[B_][GXS_];                 // h   · U
    __shared__ float s_h[B_][HG_];                   // h in f32
    __shared__ __align__(32) __bf16 s_hbf[2][2][32][16]; // h as WMMA A-frags [mi][kf][lane][slot]
    __shared__ float s_bi[N3_], s_bh[N3_];

    const int g   = blockIdx.x;
    const int tid = threadIdx.x;

    // ---- init: biases ----
    for (int i = tid; i < N3_; i += 256) {
        s_bi[i] = bi[g * N3_ + i];
        s_bh[i] = bh[g * N3_ + i];
    }

    // ---- init: h0 -> s_h (f32) and s_hbf (A-fragment layout) ----
    {
        const int b  = tid >> 3;
        const int jb = tid & 7;
        const int m  = b & 15;
        const int mi = b >> 4;
#pragma unroll
        for (int k = 0; k < 8; ++k) {
            const int j = jb + k * 8;
            const float v = h0[(g * B_ + b) * HG_ + j];
            s_h[b][j] = v;
            const int kf = j >> 5, jj = j & 31;
            const int hi = (jj >> 3) & 1;
            const int slot = (jj & 7) + ((jj >> 4) << 3);
            s_hbf[mi][kf][m + hi * 16][slot] = (__bf16)v;
        }
    }

    // ---- wave identity ----
    const int lane  = tid & 31;
    const int wid   = tid >> 5;         // 0..7
    const int mi    = wid & 1;          // which M-tile (rows 0-15 or 16-31)
    const int nbase = wid >> 1;         // N-tile base; tiles nbase, nbase+4, nbase+8
    const int half  = lane >> 4;        // K-half of the fragment held by this lane
    const int ml    = lane & 15;
    const int bRow  = mi * 16 + ml;     // batch row this lane supplies for A

    // ---- load W/U as resident bf16 B-fragments (one-time, strided) ----
    // B layout (mirror of A): lane ml holds column N=nt*16+ml; slot s -> K =
    // kf*32 + (s&7) + ((s>>3)<<4) + half*8.
    v16bf Wf[3][2], Uf[3][2];
#pragma unroll
    for (int it = 0; it < 3; ++it) {
        const int col = (nbase + it * 4) * 16 + ml;
#pragma unroll
        for (int kf = 0; kf < 2; ++kf) {
            const float* wp = Wm + (size_t)g * IG_ * N3_ + col;
            const float* up = Um + (size_t)g * HG_ * N3_ + col;
            v16bf wv, uv;
#pragma unroll
            for (int s = 0; s < 16; ++s) {
                const int K = kf * 32 + (s & 7) + ((s >> 3) << 4) + half * 8;
                wv[s] = (__bf16)wp[K * N3_];
                uv[s] = (__bf16)up[K * N3_];
            }
            Wf[it][kf] = wv;
            Uf[it][kf] = uv;
        }
    }

    __syncthreads();

    const int ewB  = tid >> 3;          // elementwise: batch row
    const int ewJb = tid & 7;
    const int ewM  = ewB & 15;
    const int ewMi = ewB >> 4;

    for (int t = 0; t < T_; ++t) {
        // ---- A-fragments: x_t (global, f32->bf16) and h (LDS, pre-laid-out) ----
        v16bf a1[2], a2[2];
#pragma unroll
        for (int kf = 0; kf < 2; ++kf) {
            const float* xr = x + ((size_t)bRow * T_ + t) * ITOT_ + g * IG_ + kf * 32 + half * 8;
            const float4 p0 = *(const float4*)(xr + 0);
            const float4 p1 = *(const float4*)(xr + 4);
            const float4 p2 = *(const float4*)(xr + 16);
            const float4 p3 = *(const float4*)(xr + 20);
            v16bf a;
            a[0]  = (__bf16)p0.x; a[1]  = (__bf16)p0.y; a[2]  = (__bf16)p0.z; a[3]  = (__bf16)p0.w;
            a[4]  = (__bf16)p1.x; a[5]  = (__bf16)p1.y; a[6]  = (__bf16)p1.z; a[7]  = (__bf16)p1.w;
            a[8]  = (__bf16)p2.x; a[9]  = (__bf16)p2.y; a[10] = (__bf16)p2.z; a[11] = (__bf16)p2.w;
            a[12] = (__bf16)p3.x; a[13] = (__bf16)p3.y; a[14] = (__bf16)p3.z; a[15] = (__bf16)p3.w;
            a1[kf] = a;
            a2[kf] = *(const v16bf*)&s_hbf[mi][kf][lane][0];
        }

        // Prefetch next timestep's x slice at WGP scope (locality=3 -> pulls
        // into all cache levels; this WGP consumes it next iteration).
        if (t + 1 < T_) {
            __builtin_prefetch(x + ((size_t)bRow * T_ + (t + 1)) * ITOT_ + g * IG_ + half * 32, 0, 3);
        }

        // ---- 12 bf16 WMMAs per wave: gx and gh tiles, K=64 in two chunks ----
#pragma unroll
        for (int it = 0; it < 3; ++it) {
            const int col = (nbase + it * 4) * 16 + ml;
            v8f cw = {};
            cw = __builtin_amdgcn_wmma_f32_16x16x32_bf16(false, a1[0], false, Wf[it][0], (short)0, cw, false, false);
            cw = __builtin_amdgcn_wmma_f32_16x16x32_bf16(false, a1[1], false, Wf[it][1], (short)0, cw, false, false);
            v8f cu = {};
            cu = __builtin_amdgcn_wmma_f32_16x16x32_bf16(false, a2[0], false, Uf[it][0], (short)0, cu, false, false);
            cu = __builtin_amdgcn_wmma_f32_16x16x32_bf16(false, a2[1], false, Uf[it][1], (short)0, cu, false, false);
#pragma unroll
            for (int v = 0; v < 8; ++v) {
                const int row = mi * 16 + half * 8 + v;   // C/D layout: VGPR v -> M=v / M=8+v
                s_gx[row][col] = cw[v];
                s_gh[row][col] = cu[v];
            }
        }

        __syncthreads();

        // ---- gates + state update (256 threads x 8 elems = 32x64) ----
        float* yrow = y + ((size_t)ewB * T_ + t) * ITOT_ + g * HG_;
#pragma unroll
        for (int k = 0; k < 8; ++k) {
            const int j = ewJb + k * 8;
            const float zs = s_gx[ewB][j]       + s_bi[j]       + s_gh[ewB][j]       + s_bh[j];
            const float rs = s_gx[ewB][64 + j]  + s_bi[64 + j]  + s_gh[ewB][64 + j]  + s_bh[64 + j];
            const float zg = 1.0f / (1.0f + __expf(-zs));
            const float rg = 1.0f / (1.0f + __expf(-rs));
            const float xn = s_gx[ewB][128 + j] + s_bi[128 + j];
            const float hn = s_gh[ewB][128 + j] + s_bh[128 + j];
            const float ng = tanhf(xn + rg * hn);
            const float hnew = zg * s_h[ewB][j] + (1.0f - zg) * ng;
            s_h[ewB][j] = hnew;
            // y is write-once / never re-read: stream it past the caches (TH=NT).
            __builtin_nontemporal_store(hnew, &yrow[j]);
            const int kf = j >> 5, jj = j & 31;
            const int hi = (jj >> 3) & 1;
            const int slot = (jj & 7) + ((jj >> 4) << 3);
            s_hbf[ewMi][kf][ewM + hi * 16][slot] = (__bf16)hnew;
        }

        __syncthreads();
    }

    // ---- final hidden state ----
#pragma unroll
    for (int k = 0; k < 8; ++k) {
        const int j = ewJb + k * 8;
        hout[(g * B_ + ewB) * HG_ + j] = s_h[ewB][j];
    }
}

extern "C" void kernel_launch(void* const* d_in, const int* in_sizes, int n_in,
                              void* d_out, int out_size, void* d_ws, size_t ws_size,
                              hipStream_t stream) {
    (void)in_sizes; (void)n_in; (void)d_ws; (void)ws_size; (void)out_size;
    const float* x  = (const float*)d_in[0];
    const float* h0 = (const float*)d_in[1];
    const float* W  = (const float*)d_in[2];
    const float* U  = (const float*)d_in[3];
    const float* bi = (const float*)d_in[4];
    const float* bh = (const float*)d_in[5];
    float* y    = (float*)d_out;                       // [B,T,512]
    float* hout = y + (size_t)B_ * T_ * ITOT_;         // [G,B,64] appended
    grouped_gru_persistent<<<dim3(G_), dim3(256), 0, stream>>>(x, h0, W, U, bi, bh, y, hout);
}
